// MultiHeadedAttention_54485955117650
// MI455X (gfx1250) — compile-verified, matrix operand order checked by reading, not run on hardware
//
#include <hip/hip_runtime.h>
#include <hip/hip_bf16.h>
#include <math.h>

typedef __attribute__((ext_vector_type(16))) __bf16 v16bf;
typedef __attribute__((ext_vector_type(8)))  __bf16 v8bf;
typedef __attribute__((ext_vector_type(2)))  __bf16 v2bf;
typedef __attribute__((ext_vector_type(8)))  float  v8f;

#define BT_   16
#define D_    256
#define H_    32
#define W_    32
#define B_    2
#define T_    8
#define NTOK  16384   // BT_*H_*W_

// ---------------------------------------------------------------------------
// Optional CDNA5 async global->LDS copy (ASYNCcnt-tracked), guarded so the
// build cannot break if the builtin is absent. Parameter types derived from
// the compiler diagnostic: (AS1 int4* src, AS3 int4* dst, imm, imm).
// flat->AS1 is an identity mapping; flat LDS address bits [31:0] are the DS
// offset (ISA 10.2 aperture rules), so integer casts are semantically right.
// ---------------------------------------------------------------------------
#define AS1 __attribute__((address_space(1)))
#define AS3 __attribute__((address_space(3)))
typedef int v4i_gcc __attribute__((vector_size(4 * sizeof(int))));

#if defined(__AMDGCN__) && __has_builtin(__builtin_amdgcn_global_load_async_to_lds_b128)
#define HAS_ASYNC_LDS 1
__device__ __forceinline__ void async_load_b128(void* lds, const void* g) {
  __builtin_amdgcn_global_load_async_to_lds_b128(
      (AS1 v4i_gcc*)(unsigned long long)g,
      (AS3 v4i_gcc*)(unsigned int)(unsigned long long)lds, 0, 0);
}
__device__ __forceinline__ void wait_async() {
#if __has_builtin(__builtin_amdgcn_s_wait_asynccnt)
  __builtin_amdgcn_s_wait_asynccnt(0);
#else
  asm volatile("s_wait_asynccnt 0x0" ::: "memory");
#endif
}
#else
#define HAS_ASYNC_LDS 0
#endif

// ---------------------------------------------------------------------------
// WMMA fragment loaders (wave32, CDNA5 ISA 7.12.2 layouts)
// A: 16x32 bf16, row M = lane&15; per-lane data is two contiguous 16B chunks.
// B_T: logical B[k][n] read from row-major M[n][k] -> same contiguous shape.
// C/D: element r of v8f = (M = r + 8*(lane/16), N = lane&15)
// ---------------------------------------------------------------------------
__device__ __forceinline__ v16bf load_A16x32(const __bf16* base, int ld) {
  const int l = threadIdx.x & 31;
  const int m = l & 15, h = l >> 4;
  v16bf a;
#pragma unroll
  for (int v = 0; v < 8; ++v) {
    const int k = (v >> 2) * 16 + h * 8 + (v & 3) * 2;
    v2bf p = *(const v2bf*)(base + m * ld + k);
    a[2 * v] = p.x; a[2 * v + 1] = p.y;
  }
  return a;
}

__device__ __forceinline__ v16bf load_B32x16_T(const __bf16* base, int ld) {
  const int l = threadIdx.x & 31;
  const int n = l & 15, h = l >> 4;
  v16bf b;
#pragma unroll
  for (int v = 0; v < 8; ++v) {
    const int k = (v >> 2) * 16 + h * 8 + (v & 3) * 2;
    v2bf p = *(const v2bf*)(base + n * ld + k);
    b[2 * v] = p.x; b[2 * v + 1] = p.y;
  }
  return b;
}

#define WMMA_BF16(A, Bm, C) \
  __builtin_amdgcn_wmma_f32_16x16x32_bf16(false, (A), false, (Bm), (short)0, (C), false, false)

// scatter one attention-output element (b, l, dcol) back into token-major attn[n][ch]
template<int CB, int PW, int PH>
__device__ __forceinline__ void scatter_y(__bf16* attn, int b, int l, int dcol, float val) {
  constexpr int OW = W_ / PW, OH = H_ / PH;
  const int t  = l / (OH * OW);
  const int rm = l % (OH * OW);
  const int oy = rm / OW, ox = rm % OW;
  const int cc = dcol / (PW * PH);
  const int pr = dcol % (PW * PH);
  const int py = pr / PW, px = pr % PW;
  const int n = ((b * T_ + t) * H_ + oy * PH + py) * W_ + ox * PW + px;
  attn[(size_t)n * D_ + CB + cc] = (__bf16)val;
}

// ---------------------------------------------------------------------------
// 1. pack: x (bt,c,h,w) fp32 -> xT [n][256] bf16 ; weights fp32 -> bf16
//    wo (oc,ic,3,3) -> wob [(ky*3+kx)][oc][ic] bf16  (B_T-friendly)
// ---------------------------------------------------------------------------
__global__ __launch_bounds__(256)
void pack_kernel(const float* __restrict__ x,
                 const float* __restrict__ wq, const float* __restrict__ wk,
                 const float* __restrict__ wv, const float* __restrict__ wo,
                 __bf16* __restrict__ xT,
                 __bf16* __restrict__ wqb, __bf16* __restrict__ wkb,
                 __bf16* __restrict__ wvb, __bf16* __restrict__ wob) {
  long idx = (long)blockIdx.x * 256 + threadIdx.x;
  const long NX = (long)NTOK * 256;
  if (idx < NX) {
    const int n = (int)(idx >> 8), c = (int)(idx & 255);
    xT[idx] = (__bf16)x[(long)(n >> 10) * (256 * 1024) + (long)c * 1024 + (n & 1023)];
    return;
  }
  idx -= NX;
  if (idx < 65536) { wqb[idx] = (__bf16)wq[idx]; return; }
  idx -= 65536;
  if (idx < 65536) { wkb[idx] = (__bf16)wk[idx]; return; }
  idx -= 65536;
  if (idx < 65536) { wvb[idx] = (__bf16)wv[idx]; return; }
  idx -= 65536;
  if (idx < 589824) {
    // idx = ((kyx*256)+oc)*256 + ic
    const int ic = (int)(idx & 255);
    const long r = idx >> 8;
    const int oc = (int)(r & 255);
    const int kyx = (int)(r >> 8);
    wob[idx] = (__bf16)wo[((long)oc * 256 + ic) * 9 + kyx];
  }
}

// ---------------------------------------------------------------------------
// 2. QKV projection GEMM: q/k/v[n][o] = sum_c xT[n][c]*W[o][c] + bias[o]
//    grid (NTOK/16, 2, 3), block 256 (8 waves x 16 cols). A tile staged in LDS
//    via async global->LDS copies (b128 chunks) when available.
// ---------------------------------------------------------------------------
__global__ __launch_bounds__(256)
void qkv_gemm_kernel(const __bf16* __restrict__ xT,
                     const __bf16* __restrict__ wqb, const __bf16* __restrict__ wkb,
                     const __bf16* __restrict__ wvb,
                     const float* __restrict__ bq, const float* __restrict__ bk,
                     const float* __restrict__ bv,
                     __bf16* __restrict__ q, __bf16* __restrict__ k,
                     __bf16* __restrict__ v) {
  __shared__ __bf16 ldsA[16 * 256];
  const __bf16* w; const float* bias; __bf16* out;
  if (blockIdx.z == 0)      { w = wqb; bias = bq; out = q; }
  else if (blockIdx.z == 1) { w = wkb; bias = bk; out = k; }
  else                      { w = wvb; bias = bv; out = v; }

  const int rowBase = blockIdx.x * 16;
  // 16 rows x 512B = 512 chunks of 16B
#if HAS_ASYNC_LDS
  for (int c = threadIdx.x; c < 512; c += 256) {
    const int row = c >> 5, col8 = (c & 31) * 8;
    async_load_b128(ldsA + row * 256 + col8,
                    xT + (size_t)(rowBase + row) * 256 + col8);
  }
  wait_async();
#else
  for (int c = threadIdx.x; c < 512; c += 256) {
    const int row = c >> 5, col8 = (c & 31) * 8;
    *(v8bf*)(ldsA + row * 256 + col8) =
        *(const v8bf*)(xT + (size_t)(rowBase + row) * 256 + col8);
  }
#endif
  __syncthreads();

  const int wave = threadIdx.x >> 5, lane = threadIdx.x & 31;
  const int n16 = lane & 15, half = lane >> 4;
  const int col0 = blockIdx.y * 128 + wave * 16;

  v8f acc = {};
#pragma unroll
  for (int kk = 0; kk < 8; ++kk) {
    v16bf a  = load_A16x32(ldsA + kk * 32, 256);
    v16bf bb = load_B32x16_T(w + (size_t)col0 * 256 + kk * 32, 256);
    acc = WMMA_BF16(a, bb, acc);
  }
#pragma unroll
  for (int r = 0; r < 8; ++r) {
    const int row = rowBase + r + half * 8;
    const int col = col0 + n16;
    out[(size_t)row * 256 + col] = (__bf16)(acc[r] + bias[col]);
  }
}

// ---------------------------------------------------------------------------
// 3. patchify: q/k [n][256] -> qs/ks [b][L][Dp] ; v -> vsT [b][Dp][L]
//    grid (4096, 3) x 256
// ---------------------------------------------------------------------------
__global__ __launch_bounds__(256)
void patchify_kernel(const __bf16* __restrict__ qb, const __bf16* __restrict__ kb,
                     const __bf16* __restrict__ vb,
                     __bf16* __restrict__ qs, __bf16* __restrict__ ks,
                     __bf16* __restrict__ vsT,
                     int Dp, int L, int pw, int ph, int ow, int oh, int cbase) {
  const int idx = blockIdx.x * 256 + threadIdx.x;     // < 2*L*Dp == 1048576
  const __bf16* src; __bf16* dst;
  if (blockIdx.y == 0)      { src = qb; dst = qs; }
  else if (blockIdx.y == 1) { src = kb; dst = ks; }
  else                      { src = vb; dst = vsT; }

  const int d = idx % Dp;
  const int l = (idx / Dp) % L;
  const int b = idx / (Dp * L);
  const int ox = l % ow, oy = (l / ow) % oh, t = l / (ow * oh);
  const int px = d % pw, py = (d / pw) % ph, cc = d / (pw * ph);
  const int n = ((b * T_ + t) * H_ + oy * ph + py) * W_ + ox * pw + px;
  const __bf16 val = src[(size_t)n * 256 + cbase + cc];
  if (blockIdx.y == 2) dst[((size_t)b * Dp + d) * L + l] = val;   // transposed
  else                 dst[idx] = val;
}

// ---------------------------------------------------------------------------
// 4a. scores GEMM (scales 0,1):  S[b][l][m] = (qs . ks) / sqrt(Dp)
//     grid (L/16, L/64, 2), block 128
// ---------------------------------------------------------------------------
__global__ __launch_bounds__(128)
void scores_kernel(const __bf16* __restrict__ qs, const __bf16* __restrict__ ks,
                   float* __restrict__ S, int L, int Dp) {
  const int wave = threadIdx.x >> 5, lane = threadIdx.x & 31;
  const int n16 = lane & 15, half = lane >> 4;
  const int b = blockIdx.z;
  const int row0 = blockIdx.x * 16;
  const int col0 = (blockIdx.y * 4 + wave) * 16;
  const __bf16* qsb = qs + (size_t)b * L * Dp;
  const __bf16* ksb = ks + (size_t)b * L * Dp;

  v8f acc = {};
  for (int kk = 0; kk < Dp / 32; ++kk) {
    v16bf a  = load_A16x32(qsb + (size_t)row0 * Dp + kk * 32, Dp);
    v16bf bb = load_B32x16_T(ksb + (size_t)col0 * Dp + kk * 32, Dp);
    acc = WMMA_BF16(a, bb, acc);
  }
  const float sc = rsqrtf((float)Dp);
#pragma unroll
  for (int r = 0; r < 8; ++r)
    S[((size_t)b * L + row0 + r + half * 8) * L + col0 + n16] = acc[r] * sc;
}

// ---------------------------------------------------------------------------
// 4b. softmax over rows of S -> P (bf16).  grid (2*L), block 256
// ---------------------------------------------------------------------------
__global__ __launch_bounds__(256)
void softmax_kernel(const float* __restrict__ S, __bf16* __restrict__ P, int L) {
  __shared__ float red[256];
  const int row = blockIdx.x;
  const float* Sr = S + (size_t)row * L;
  __bf16* Pr = P + (size_t)row * L;
  const int tid = threadIdx.x;

  float mx = -INFINITY;
  for (int i = tid; i < L; i += 256) mx = fmaxf(mx, Sr[i]);
  red[tid] = mx; __syncthreads();
  for (int s = 128; s > 0; s >>= 1) {
    if (tid < s) red[tid] = fmaxf(red[tid], red[tid + s]);
    __syncthreads();
  }
  mx = red[0]; __syncthreads();

  float sum = 0.f;
  for (int i = tid; i < L; i += 256) sum += __expf(Sr[i] - mx);
  red[tid] = sum; __syncthreads();
  for (int s = 128; s > 0; s >>= 1) {
    if (tid < s) red[tid] += red[tid + s];
    __syncthreads();
  }
  const float inv = 1.0f / red[0];
  for (int i = tid; i < L; i += 256) Pr[i] = (__bf16)(__expf(Sr[i] - mx) * inv);
}

// ---------------------------------------------------------------------------
// 4c. PV GEMM (scales 0,1): Y = P * V (V transposed layout), scatter into attn.
//     grid (L/16, DP/64, 2), block 128
// ---------------------------------------------------------------------------
template<int DP, int CB, int PW, int PH>
__global__ __launch_bounds__(128)
void pv_kernel(const __bf16* __restrict__ P, const __bf16* __restrict__ vsT,
               __bf16* __restrict__ attn, int L) {
  const int wave = threadIdx.x >> 5, lane = threadIdx.x & 31;
  const int n16 = lane & 15, half = lane >> 4;
  const int b = blockIdx.z;
  const int row0 = blockIdx.x * 16;
  const int col0 = (blockIdx.y * 4 + wave) * 16;
  const __bf16* Pb  = P   + (size_t)b * L * L;
  const __bf16* vtb = vsT + (size_t)b * DP * L;

  v8f acc = {};
  for (int kk = 0; kk < L / 32; ++kk) {
    v16bf a  = load_A16x32(Pb + (size_t)row0 * L + kk * 32, L);
    v16bf bb = load_B32x16_T(vtb + (size_t)col0 * L + kk * 32, L);
    acc = WMMA_BF16(a, bb, acc);
  }
#pragma unroll
  for (int r = 0; r < 8; ++r)
    scatter_y<CB, PW, PH>(attn, b, row0 + r + half * 8, col0 + n16, acc[r]);
}

// ---------------------------------------------------------------------------
// 5. flash attention (scales 2,3): one wave owns 16 query rows; Q fragments
//    hoisted into registers; online softmax; P re-laid to A-frag via LDS.
//    grid (L/64, 2), block 128 (4 waves).
// ---------------------------------------------------------------------------
template<int DP, int CB, int PW, int PH>
__global__ __launch_bounds__(128)
void flash_kernel(const __bf16* __restrict__ qs, const __bf16* __restrict__ ks,
                  const __bf16* __restrict__ vsT, __bf16* __restrict__ attn, int L) {
  constexpr int NT = DP / 16;   // output col tiles
  constexpr int KK = DP / 32;   // k-steps for QK^T
  __shared__ __bf16 ldsP[4][16 * 32];

  const int wave = threadIdx.x >> 5, lane = threadIdx.x & 31;
  const int n16 = lane & 15, half = lane >> 4;
  const int b = blockIdx.y;
  const int row0 = (blockIdx.x * 4 + wave) * 16;
  const __bf16* qsb = qs  + (size_t)b * L * DP;
  const __bf16* ksb = ks  + (size_t)b * L * DP;
  const __bf16* vtb = vsT + (size_t)b * DP * L;
  const float sc = rsqrtf((float)DP);
  __bf16* Pw = &ldsP[wave][0];

  // hoist Q fragments (reused for every key tile)
  v16bf qfrag[KK];
#pragma unroll
  for (int kk = 0; kk < KK; ++kk)
    qfrag[kk] = load_A16x32(qsb + (size_t)row0 * DP + kk * 32, DP);

  v8f zero = {};
  v8f acc[NT];
#pragma unroll
  for (int d = 0; d < NT; ++d) acc[d] = zero;
  float mrow[8], lrow[8];
#pragma unroll
  for (int r = 0; r < 8; ++r) { mrow[r] = -INFINITY; lrow[r] = 0.f; }

  for (int j = 0; j < L / 32; ++j) {
    if (j + 1 < L / 32) {   // prefetch next K tile (global_prefetch_b8)
      __builtin_prefetch(ksb + (size_t)(j + 1) * 32 * DP, 0, 1);
      __builtin_prefetch(vtb + (size_t)(j + 1) * 32, 0, 1);
    }
    v8f s0 = {}, s1 = {};
#pragma unroll
    for (int kk = 0; kk < KK; ++kk) {
      v16bf b0 = load_B32x16_T(ksb + (size_t)(j * 32)      * DP + kk * 32, DP);
      v16bf b1 = load_B32x16_T(ksb + (size_t)(j * 32 + 16) * DP + kk * 32, DP);
      s0 = WMMA_BF16(qfrag[kk], b0, s0);
      s1 = WMMA_BF16(qfrag[kk], b1, s1);
    }
    // online softmax per row (rows live across 16 lanes of each half-wave)
#pragma unroll
    for (int r = 0; r < 8; ++r) {
      const float v0 = s0[r] * sc, v1 = s1[r] * sc;
      float mx = fmaxf(v0, v1);
      mx = fmaxf(mx, __shfl_xor(mx, 1));
      mx = fmaxf(mx, __shfl_xor(mx, 2));
      mx = fmaxf(mx, __shfl_xor(mx, 4));
      mx = fmaxf(mx, __shfl_xor(mx, 8));
      const float mnew = fmaxf(mrow[r], mx);
      const float alpha = __expf(mrow[r] - mnew);
      const float p0 = __expf(v0 - mnew), p1 = __expf(v1 - mnew);
      float rs = p0 + p1;
      rs += __shfl_xor(rs, 1);
      rs += __shfl_xor(rs, 2);
      rs += __shfl_xor(rs, 4);
      rs += __shfl_xor(rs, 8);
      lrow[r] = lrow[r] * alpha + rs;
      mrow[r] = mnew;
#pragma unroll
      for (int d = 0; d < NT; ++d) acc[d][r] *= alpha;
      const int M = r + half * 8;
      Pw[M * 32 + n16]      = (__bf16)p0;
      Pw[M * 32 + 16 + n16] = (__bf16)p1;
    }
    // P(16x32) * V(32xDP)   (per-wave private LDS; DS pipeline is in-order)
    v16bf pa = load_A16x32(Pw, 32);
#pragma unroll
    for (int d = 0; d < NT; ++d) {
      v16bf bb = load_B32x16_T(vtb + (size_t)(d * 16) * L + j * 32, L);
      acc[d] = WMMA_BF16(pa, bb, acc[d]);
    }
  }
#pragma unroll
  for (int r = 0; r < 8; ++r) {
    const float inv = 1.0f / lrow[r];
    const int l = row0 + r + half * 8;
#pragma unroll
    for (int d = 0; d < NT; ++d)
      scatter_y<CB, PW, PH>(attn, b, l, d * 16 + n16, acc[d][r] * inv);
  }
}

// ---------------------------------------------------------------------------
// 6. 3x3 conv as implicit GEMM + bias + LeakyReLU(0.2).  grid (1024, 4) x 128
//    Input tile staged via async global->LDS (zero-filled border rows).
// ---------------------------------------------------------------------------
__global__ __launch_bounds__(128)
void conv_kernel(const __bf16* __restrict__ attn, const __bf16* __restrict__ wob,
                 const float* __restrict__ bo, float* __restrict__ out) {
  __shared__ __bf16 ldsA[16 * 32];
  const int rowBase = blockIdx.x * 16;         // 16 tokens, same (bt, y), x aligned
  const int wave = threadIdx.x >> 5, lane = threadIdx.x & 31;
  const int n16 = lane & 15, half = lane >> 4;
  const int oc0 = blockIdx.y * 64 + wave * 16;
  const int bt = rowBase >> 10;
  const int ybase = (rowBase >> 5) & 31;
  const int xbase = rowBase & 31;              // 0 or 16

  v8f acc = {};
  for (int ky = 0; ky < 3; ++ky)
    for (int kx = 0; kx < 3; ++kx)
      for (int kc = 0; kc < 8; ++kc) {
        // stage 16 rows x 64B; one 16B chunk per thread (threads 0..63)
        if (threadIdx.x < 64) {
          const int tr = threadIdx.x >> 2, c = threadIdx.x & 3;
          const int yy = ybase + ky - 1;
          const int xx = xbase + tr + kx - 1;
          __bf16* ldst = ldsA + tr * 32 + c * 8;
          const bool inb = (yy >= 0 && yy < 32 && xx >= 0 && xx < 32);
          const __bf16* g =
              attn + (size_t)((bt * 32 + yy) * 32 + xx) * 256 + kc * 32 + c * 8;
#if HAS_ASYNC_LDS
          if (inb) async_load_b128(ldst, g);
          else     { v8bf z = {}; *(v8bf*)ldst = z; }
#else
          if (inb) *(v8bf*)ldst = *(const v8bf*)g;
          else     { v8bf z = {}; *(v8bf*)ldst = z; }
#endif
        }
#if HAS_ASYNC_LDS
        wait_async();
#endif
        __syncthreads();
        v16bf a  = load_A16x32(ldsA, 32);
        v16bf bb = load_B32x16_T(
            wob + (size_t)(ky * 3 + kx) * 65536 + (size_t)oc0 * 256 + kc * 32, 256);
        acc = WMMA_BF16(a, bb, acc);
        __syncthreads();
      }
#pragma unroll
  for (int r = 0; r < 8; ++r) {
    const int n = rowBase + r + half * 8;
    const int oc = oc0 + n16;
    float vv = acc[r] + bo[oc];
    vv = vv > 0.f ? vv : 0.2f * vv;
    out[(size_t)(bt * 256 + oc) * 1024 + (n & 1023)] = vv;
  }
}

// ---------------------------------------------------------------------------
extern "C" void kernel_launch(void* const* d_in, const int* in_sizes, int n_in,
                              void* d_out, int out_size, void* d_ws, size_t ws_size,
                              hipStream_t stream) {
  const float* x  = (const float*)d_in[0];
  const float* wq = (const float*)d_in[2];
  const float* bq = (const float*)d_in[3];
  const float* wk = (const float*)d_in[4];
  const float* bk = (const float*)d_in[5];
  const float* wv = (const float*)d_in[6];
  const float* bv = (const float*)d_in[7];
  const float* wo = (const float*)d_in[8];
  const float* bo = (const float*)d_in[9];
  float* out = (float*)d_out;

  char* wsp = (char*)d_ws;
  auto alloc = [&](size_t bytes) {
    void* p = (void*)wsp;
    wsp += (bytes + 255) & ~(size_t)255;
    return p;
  };
  __bf16* xT   = (__bf16*)alloc((size_t)NTOK * 256 * 2);
  __bf16* qb   = (__bf16*)alloc((size_t)NTOK * 256 * 2);
  __bf16* kb   = (__bf16*)alloc((size_t)NTOK * 256 * 2);
  __bf16* vb   = (__bf16*)alloc((size_t)NTOK * 256 * 2);
  __bf16* attn = (__bf16*)alloc((size_t)NTOK * 256 * 2);
  __bf16* wqb  = (__bf16*)alloc(65536 * 2);
  __bf16* wkb  = (__bf16*)alloc(65536 * 2);
  __bf16* wvb  = (__bf16*)alloc(65536 * 2);
  __bf16* wob  = (__bf16*)alloc(589824 * 2);
  __bf16* qs   = (__bf16*)alloc(1048576 * 2);
  __bf16* ks   = (__bf16*)alloc(1048576 * 2);
  __bf16* vsT  = (__bf16*)alloc(1048576 * 2);
  float*  S    = (float*) alloc((size_t)2 * 512 * 512 * 4);
  __bf16* P    = (__bf16*)alloc((size_t)2 * 512 * 512 * 2);

  pack_kernel<<<19456, 256, 0, stream>>>(x, wq, wk, wv, wo, xT, wqb, wkb, wvb, wob);

  qkv_gemm_kernel<<<dim3(NTOK / 16, 2, 3), 256, 0, stream>>>(
      xT, wqb, wkb, wvb, bq, bk, bv, qb, kb, vb);

  // ---- scale 0: patch 8x8, L=128, Dp=4096 ----
  patchify_kernel<<<dim3(4096, 3), 256, 0, stream>>>(qb, kb, vb, qs, ks, vsT,
                                                     4096, 128, 8, 8, 4, 4, 0);
  scores_kernel<<<dim3(128 / 16, 128 / 64, 2), 128, 0, stream>>>(qs, ks, S, 128, 4096);
  softmax_kernel<<<2 * 128, 256, 0, stream>>>(S, P, 128);
  pv_kernel<4096, 0, 8, 8><<<dim3(128 / 16, 4096 / 64, 2), 128, 0, stream>>>(P, vsT, attn, 128);

  // ---- scale 1: patch 4x4, L=512, Dp=1024 ----
  patchify_kernel<<<dim3(4096, 3), 256, 0, stream>>>(qb, kb, vb, qs, ks, vsT,
                                                     1024, 512, 4, 4, 8, 8, 64);
  scores_kernel<<<dim3(512 / 16, 512 / 64, 2), 128, 0, stream>>>(qs, ks, S, 512, 1024);
  softmax_kernel<<<2 * 512, 256, 0, stream>>>(S, P, 512);
  pv_kernel<1024, 64, 4, 4><<<dim3(512 / 16, 1024 / 64, 2), 128, 0, stream>>>(P, vsT, attn, 512);

  // ---- scale 2: patch 2x2, L=2048, Dp=256 (flash) ----
  patchify_kernel<<<dim3(4096, 3), 256, 0, stream>>>(qb, kb, vb, qs, ks, vsT,
                                                     256, 2048, 2, 2, 16, 16, 128);
  flash_kernel<256, 128, 2, 2><<<dim3(2048 / 64, 2), 128, 0, stream>>>(qs, ks, vsT, attn, 2048);

  // ---- scale 3: patch 1x1, L=8192, Dp=64 (flash) ----
  patchify_kernel<<<dim3(4096, 3), 256, 0, stream>>>(qb, kb, vb, qs, ks, vsT,
                                                     64, 8192, 1, 1, 32, 32, 192);
  flash_kernel<64, 192, 1, 1><<<dim3(8192 / 64, 2), 128, 0, stream>>>(qs, ks, vsT, attn, 8192);

  // ---- 3x3 conv + bias + LeakyReLU ----
  conv_kernel<<<dim3(NTOK / 16, 4), 128, 0, stream>>>(attn, wob, bo, out);
}
